// BatchMultiHeadGraphAttention_1872605741371
// MI455X (gfx1250) — matrix operand order repaired and checked by reading, not compiled, not run on hardware
//
#include <hip/hip_runtime.h>
#include <hip/hip_bf16.h>

typedef __attribute__((ext_vector_type(16))) _Float16 v16h;
typedef __attribute__((ext_vector_type(8)))  float    v8f;
typedef __attribute__((ext_vector_type(4)))  float    v4f;

#define NEG_SLOPE 0.2f
#define BS 8
#define NH 4
#define NBH 32          // BS*NH
#define N 2048
#define FIN 64
#define FOUT 32

__device__ __forceinline__ float lrelu(float x) {
    return x >= 0.0f ? x : NEG_SLOPE * x;
}

// ---------------------------------------------------------------------------
// Kernel A: h_prime^T stored f16 [bh][o][j] (WMMA B operand feed),
//           src[bh][j], dst[bh][j] in fp32.
// One wave per (bh, row); lane = output feature o.
// ---------------------------------------------------------------------------
__global__ void gat_prep(const float* __restrict__ h, const float* __restrict__ w,
                         const float* __restrict__ a_src, const float* __restrict__ a_dst,
                         _Float16* __restrict__ hpT, float* __restrict__ src,
                         float* __restrict__ dst) {
    const int bh   = blockIdx.x;            // 0..31
    const int b    = bh >> 2;
    const int head = bh & 3;
    const int wave = threadIdx.x >> 5;
    const int lane = threadIdx.x & 31;
    const int row  = blockIdx.y * 8 + wave; // 0..2047

    const float* hrow = h + ((size_t)b * N + row) * FIN;
    const float* wcol = w + head * FIN * FOUT + lane;   // stride FOUT

    float acc = 0.0f;
#pragma unroll
    for (int f = 0; f < FIN; ++f)
        acc = fmaf(hrow[f], wcol[f * FOUT], acc);

    hpT[(size_t)bh * FOUT * N + (size_t)lane * N + row] = (_Float16)acc;

    float ts = acc * a_src[head * FOUT + lane];
    float td = acc * a_dst[head * FOUT + lane];
#pragma unroll
    for (int off = 16; off; off >>= 1) {
        ts += __shfl_xor(ts, off, 32);
        td += __shfl_xor(td, off, 32);
    }
    if (lane == 0) {
        src[bh * N + row] = ts;
        dst[bh * N + row] = td;
    }
}

// ---------------------------------------------------------------------------
// Kernel B: per-row softmax max (via lrelu monotonicity) and 1/denominator.
// One wave per row; lanes split j.
// ---------------------------------------------------------------------------
__global__ void gat_denom(const float* __restrict__ src, const float* __restrict__ dst,
                          float* __restrict__ rowm, float* __restrict__ invden) {
    const int bh   = blockIdx.x;
    const int wave = threadIdx.x >> 5;
    const int lane = threadIdx.x & 31;
    const int row  = blockIdx.y * 8 + wave;

    const float* d = dst + bh * N;
    float mx = -3.0e38f;
    for (int j = lane; j < N; j += 32) mx = fmaxf(mx, d[j]);
#pragma unroll
    for (int off = 16; off; off >>= 1) mx = fmaxf(mx, __shfl_xor(mx, off, 32));

    const float s = src[bh * N + row];
    const float m = lrelu(s + mx);          // == max_j lrelu(s + dst_j)

    float sum = 0.0f;
    for (int j = lane; j < N; j += 32)
        sum += __expf(lrelu(s + d[j]) - m);
#pragma unroll
    for (int off = 16; off; off >>= 1) sum += __shfl_xor(sum, off, 32);

    if (lane == 0) {
        rowm[bh * N + row]   = m;
        invden[bh * N + row] = 1.0f / sum;
    }
}

// ---------------------------------------------------------------------------
// Kernel C: per wave, 32 rows x 32 cols of output via WMMA, sweeping j.
// A operand (p~) is computed directly in registers, contiguous-K order:
//   lane m (m=lane&15), kb=(lane>>4)*16: elements 0..15 -> K=kb+e.
// B operand layout (contiguous-K convention, same order as A):
//   lanes 0-15: elements 0..15 -> K=0..15; lanes 16-31: K=16..31,
//   i.e. one contiguous v16h load from f16 hpT.
// attn is streamed out with non-temporal fp32 b128 stores.
// Block = 256 threads (8 waves); grid = (32 bh, 8 row-blocks of 256).
// ---------------------------------------------------------------------------
__global__ void gat_main(const _Float16* __restrict__ hpT, const float* __restrict__ src,
                         const float* __restrict__ dst, const float* __restrict__ rowm,
                         const float* __restrict__ invden, const float* __restrict__ bias,
                         float* __restrict__ outp, float* __restrict__ attn) {
    const int bh   = blockIdx.x;
    const int wave = threadIdx.x >> 5;
    const int lane = threadIdx.x & 31;
    const int i0   = blockIdx.y * 256 + wave * 32;  // this wave: rows i0 .. i0+31

    const int m   = lane & 15;          // A row within tile
    const int kb  = (lane >> 4) * 16;   // A-operand K base (contiguous 16)
    const int kbB = (lane >> 4) * 16;   // B-operand K base (contiguous 16)

    const _Float16* hpTb = hpT + (size_t)bh * FOUT * N;
    const float*    dstb = dst + bh * N;

    // per-row scalars for the two A (row) tiles
    const float s0  = src[bh * N + i0 + m];
    const float s1  = src[bh * N + i0 + 16 + m];
    const float rm0 = rowm[bh * N + i0 + m];
    const float rm1 = rowm[bh * N + i0 + 16 + m];
    const float id0 = invden[bh * N + i0 + m];
    const float id1 = invden[bh * N + i0 + 16 + m];

    float* attnRow0 = attn + ((size_t)(bh * N + i0 + m)) * N;
    float* attnRow1 = attn + ((size_t)(bh * N + i0 + 16 + m)) * N;

    // B column pointers (two N-tiles: cols 0..15 and 16..31)
    const _Float16* bp0 = hpTb + (size_t)(lane & 15) * N;
    const _Float16* bp1 = bp0 + (size_t)16 * N;

    v8f c00 = {}, c01 = {}, c10 = {}, c11 = {};

    for (int j0 = 0; j0 < N; j0 += 32) {
        // dst values this lane needs for A: K = kb+0..15
        v4f dA = *(const v4f*)(dstb + j0 + kb);
        v4f dB = *(const v4f*)(dstb + j0 + kb + 4);
        v4f dC = *(const v4f*)(dstb + j0 + kb + 8);
        v4f dD = *(const v4f*)(dstb + j0 + kb + 12);
        float dv[16];
#pragma unroll
        for (int e = 0; e < 4; ++e) {
            dv[e] = dA[e]; dv[4 + e] = dB[e]; dv[8 + e] = dC[e]; dv[12 + e] = dD[e];
        }

        v16h A0, A1;
        float p0[16], p1[16];
#pragma unroll
        for (int e = 0; e < 16; ++e) {
            const float dj = dv[e];
            p0[e] = __expf(lrelu(s0 + dj) - rm0);
            p1[e] = __expf(lrelu(s1 + dj) - rm1);
            A0[e] = (_Float16)p0[e];
            A1[e] = (_Float16)p1[e];
        }

        // stream normalized attention rows (fp32, non-temporal)
        v4f w0a = { p0[0] * id0, p0[1] * id0, p0[2] * id0, p0[3] * id0 };
        v4f w0b = { p0[4] * id0, p0[5] * id0, p0[6] * id0, p0[7] * id0 };
        v4f w0c = { p0[8] * id0, p0[9] * id0, p0[10] * id0, p0[11] * id0 };
        v4f w0d = { p0[12] * id0, p0[13] * id0, p0[14] * id0, p0[15] * id0 };
        v4f w1a = { p1[0] * id1, p1[1] * id1, p1[2] * id1, p1[3] * id1 };
        v4f w1b = { p1[4] * id1, p1[5] * id1, p1[6] * id1, p1[7] * id1 };
        v4f w1c = { p1[8] * id1, p1[9] * id1, p1[10] * id1, p1[11] * id1 };
        v4f w1d = { p1[12] * id1, p1[13] * id1, p1[14] * id1, p1[15] * id1 };
        __builtin_nontemporal_store(w0a, (v4f*)(attnRow0 + j0 + kb));
        __builtin_nontemporal_store(w0b, (v4f*)(attnRow0 + j0 + kb + 4));
        __builtin_nontemporal_store(w0c, (v4f*)(attnRow0 + j0 + kb + 8));
        __builtin_nontemporal_store(w0d, (v4f*)(attnRow0 + j0 + kb + 12));
        __builtin_nontemporal_store(w1a, (v4f*)(attnRow1 + j0 + kb));
        __builtin_nontemporal_store(w1b, (v4f*)(attnRow1 + j0 + kb + 4));
        __builtin_nontemporal_store(w1c, (v4f*)(attnRow1 + j0 + kb + 8));
        __builtin_nontemporal_store(w1d, (v4f*)(attnRow1 + j0 + kb + 12));

        // B operands: 16 consecutive K values per lane, straight f16 loads
        v16h B0 = *(const v16h*)(bp0 + j0 + kbB);
        v16h B1 = *(const v16h*)(bp1 + j0 + kbB);

        c00 = __builtin_amdgcn_wmma_f32_16x16x32_f16(false, A0, false, B0, (short)0, c00, false, false);
        c01 = __builtin_amdgcn_wmma_f32_16x16x32_f16(false, A0, false, B1, (short)0, c01, false, false);
        c10 = __builtin_amdgcn_wmma_f32_16x16x32_f16(false, A1, false, B0, (short)0, c10, false, false);
        c11 = __builtin_amdgcn_wmma_f32_16x16x32_f16(false, A1, false, B1, (short)0, c11, false, false);
    }

    // Epilogue. C/D layout: lanes 0-15 -> M = v, lanes 16-31 -> M = v+8; N = lane&15.
    const int crow = (lane >> 4) * 8;
    const int ccol = lane & 15;
#pragma unroll
    for (int v = 0; v < 8; ++v) {
        const int r0 = i0 + crow + v;
        const int r1 = i0 + 16 + crow + v;
        const float idr0 = invden[bh * N + r0];
        const float idr1 = invden[bh * N + r1];
        const float bc0 = bias[ccol];
        const float bc1 = bias[ccol + 16];
        outp[((size_t)(bh * N + r0)) * FOUT + ccol]      = c00[v] * idr0 + bc0;
        outp[((size_t)(bh * N + r0)) * FOUT + ccol + 16] = c01[v] * idr0 + bc1;
        outp[((size_t)(bh * N + r1)) * FOUT + ccol]      = c10[v] * idr1 + bc0;
        outp[((size_t)(bh * N + r1)) * FOUT + ccol + 16] = c11[v] * idr1 + bc1;
    }
}

// ---------------------------------------------------------------------------
extern "C" void kernel_launch(void* const* d_in, const int* in_sizes, int n_in,
                              void* d_out, int out_size, void* d_ws, size_t ws_size,
                              hipStream_t stream) {
    const float* h     = (const float*)d_in[0];  // [8,2048,64]
    const float* w     = (const float*)d_in[1];  // [4,64,32]
    const float* a_src = (const float*)d_in[2];  // [4,32,1]
    const float* a_dst = (const float*)d_in[3];  // [4,32,1]
    const float* bias  = (const float*)d_in[4];  // [32]

    float* outp = (float*)d_out;                        // [8,4,2048,32]
    float* attn = outp + (size_t)NBH * N * FOUT;        // [8,4,2048,2048]

    _Float16* hpT = (_Float16*)d_ws;                    // [32][32][2048] f16 (4 MB)
    float* src    = (float*)((char*)d_ws + (size_t)NBH * FOUT * N * sizeof(_Float16));
    float* dstv   = src + (size_t)NBH * N;
    float* rowm   = dstv + (size_t)NBH * N;
    float* invden = rowm + (size_t)NBH * N;

    dim3 gA(NBH, N / 8);
    gat_prep<<<gA, 256, 0, stream>>>(h, w, a_src, a_dst, hpT, src, dstv);

    dim3 gB(NBH, N / 8);
    gat_denom<<<gB, 256, 0, stream>>>(src, dstv, rowm, invden);

    dim3 gC(NBH, N / 256);
    gat_main<<<gC, 256, 0, stream>>>(hpT, src, dstv, rowm, invden, bias, outp, attn);
}
